// ImplicitMotionAlignment_82265803588078
// MI455X (gfx1250) — matrix operand order repaired and checked by reading, not compiled, not run on hardware
//
#include <hip/hip_runtime.h>
#include <hip/hip_bf16.h>
#include <math.h>

typedef _Float16 f16;
typedef __attribute__((ext_vector_type(16))) _Float16 v16h;
typedef __attribute__((ext_vector_type(8)))  float    v8f;

#define S_TOK  4096
#define CDIM   256
#define DQDIM  256
#define NHEADS 8
#define DHEAD  32
#define MLPDIM 1024

// ---------------------------------------------------------------------------
// WMMA helpers (CDNA5 / gfx1250, wave32)
// ---------------------------------------------------------------------------
static __device__ __forceinline__ v8f wmma_f16(v16h a, v16h b, v8f c) {
  // D(f32 16x16) = A(f16 16x32) * B(f16 32x16) + C
  return __builtin_amdgcn_wmma_f32_16x16x32_f16(false, a, false, b,
                                                (short)0, c, false, false);
}

static __device__ __forceinline__ v8f zero8() {
  v8f z = {0.f, 0.f, 0.f, 0.f, 0.f, 0.f, 0.f, 0.f};
  return z;
}

// A-matrix fragment (16x32 f16, M x K), source row-major with leading dim ld.
// lanes 0-15 row=lane, K 0-15; lanes 16-31 K 16-31 (matches B fragment order).
static __device__ __forceinline__ v16h load_a_frag(const f16* base, int ld) {
  const int lane = threadIdx.x & 31;
  const int r    = lane & 15;
  const int hi   = lane >> 4;
  const f16* p   = base + (size_t)r * ld + hi * 16;
  v16h a;
#pragma unroll
  for (int v = 0; v < 8; ++v) {
    const int k = 2 * v; // 0,2,...,14
    a[2 * v]     = p[k];
    a[2 * v + 1] = p[k + 1];
  }
  return a;
}

// B-matrix fragment (32x16 f16, K x N) where B[k][n] = src[n][k]
// (src row-major [N][K], leading dim ld) -- i.e. B = src^T, used for X*W^T
// and Q*K^T. Lanes 0-15 hold K 0-15 (2/VGPR), lanes 16-31 hold K 16-31.
static __device__ __forceinline__ v16h load_b_frag_nk(const f16* src, int ld) {
  const int lane = threadIdx.x & 31;
  const int n    = lane & 15;
  const int kb   = (lane >> 4) * 16;
  const f16* p   = src + (size_t)n * ld + kb;
  v16h b;
#pragma unroll
  for (int v = 0; v < 8; ++v) {
    b[2 * v]     = p[2 * v];
    b[2 * v + 1] = p[2 * v + 1];
  }
  return b;
}

// B-matrix fragment (32x16 f16, K x N) from row-major [K][N] source (used for
// P*V where V tile is stored row-major keys x cols).
static __device__ __forceinline__ v16h load_b_frag_kn(const f16* src, int ld) {
  const int lane = threadIdx.x & 31;
  const int n    = lane & 15;
  const int kb   = (lane >> 4) * 16;
  const f16* p   = src + (size_t)kb * ld + n;
  v16h b;
#pragma unroll
  for (int v = 0; v < 8; ++v) {
    b[2 * v]     = p[(size_t)(2 * v) * ld];
    b[2 * v + 1] = p[(size_t)(2 * v + 1) * ld];
  }
  return b;
}

// ---------------------------------------------------------------------------
// Small prep kernels
// ---------------------------------------------------------------------------
__global__ void __launch_bounds__(256)
f32_to_f16_kernel(const float* __restrict__ src, f16* __restrict__ dst, int n) {
  int i = blockIdx.x * blockDim.x + threadIdx.x;
  if (i < n) dst[i] = (f16)src[i];
}

// q[s,d] = ml_c[d,s] + qpos[s,d] ; k likewise ; v[s,c] = fl_r[c,s]
__global__ void __launch_bounds__(256)
build_qkv_kernel(const float* __restrict__ mlc, const float* __restrict__ mlr,
                 const float* __restrict__ flr, const float* __restrict__ qpos,
                 const float* __restrict__ kpos, f16* __restrict__ q,
                 f16* __restrict__ k, f16* __restrict__ v) {
  const int s = blockIdx.x;
  const int d = threadIdx.x;
  const size_t sd = (size_t)s * DQDIM + d;
  const size_t ds = (size_t)d * S_TOK + s;
  q[sd] = (f16)(mlc[ds] + qpos[sd]);
  k[sd] = (f16)(mlr[ds] + kpos[sd]);
  v[sd] = (f16)(flr[ds]);
}

// out[c,s] = x[s,c]
__global__ void __launch_bounds__(256)
transpose_out_kernel(const float* __restrict__ x, float* __restrict__ out) {
  const int s = blockIdx.x;
  const int c = threadIdx.x;
  out[(size_t)c * S_TOK + s] = x[(size_t)s * CDIM + c];
}

// LayerNorm over C=256 (one row per block of 256 threads). Optionally emits
// f16(xn + qpos), f16(xn + kpos), f16(xn).
__global__ void __launch_bounds__(256)
ln_kernel(const float* __restrict__ x, const float* __restrict__ w,
          const float* __restrict__ b, const float* __restrict__ qp,
          const float* __restrict__ kp, f16* __restrict__ oq,
          f16* __restrict__ ok, f16* __restrict__ ov) {
  __shared__ float red[256];
  const int row = blockIdx.x, tid = threadIdx.x;
  const size_t base = (size_t)row * CDIM + tid;
  const float v = x[base];
  red[tid] = v;
  __syncthreads();
  for (int s = 128; s > 0; s >>= 1) {
    if (tid < s) red[tid] += red[tid + s];
    __syncthreads();
  }
  const float mu = red[0] * (1.f / CDIM);
  __syncthreads();
  const float d = v - mu;
  red[tid] = d * d;
  __syncthreads();
  for (int s = 128; s > 0; s >>= 1) {
    if (tid < s) red[tid] += red[tid + s];
    __syncthreads();
  }
  const float rstd = rsqrtf(red[0] * (1.f / CDIM) + 1e-5f);
  const float xn = d * rstd * w[tid] + b[tid];
  if (ov) ov[base] = (f16)xn;
  if (oq) oq[base] = (f16)(xn + qp[base]);
  if (ok) ok[base] = (f16)(xn + kp[base]);
}

// ---------------------------------------------------------------------------
// Flash attention: 4 waves / WG, each wave owns 16 query rows, key tile = 32.
// Head strides support per-head slices of [S, C] arrays.  The V/output column
// range is additionally split over blockIdx.z (vcol0 = z*DV) so that the
// dense cross-attention (DV total = 256) only needs DV=128 f32 accumulators
// per wave.  __launch_bounds__(128, 1) lifts the VGPR budget so Q fragments
// and accumulators stay register-resident (no scratch spills).
// ---------------------------------------------------------------------------
template <int D, int DV>
__global__ void __launch_bounds__(128, 1)
flash_attn_kernel(const f16* __restrict__ Q, const f16* __restrict__ K,
                  const f16* __restrict__ V, float* __restrict__ Of32,
                  f16* __restrict__ Of16, int ldq, int ldk, int ldv, int ldo,
                  int hsQ, int hsK, int hsV, int hsO, float scale) {
  __shared__ __align__(16) f16 lds_k[32 * D];
  __shared__ __align__(16) f16 lds_v[32 * DV];
  __shared__ __align__(16) f16 lds_p[4][16 * 32];

  const int tid  = threadIdx.x;
  const int wid  = tid >> 5;
  const int lane = tid & 31;
  const int hi   = lane >> 4;

  const f16* Qh = Q + (size_t)blockIdx.y * hsQ;
  const f16* Kh = K + (size_t)blockIdx.y * hsK;
  const f16* Vh = V + (size_t)blockIdx.y * hsV;
  const int vcol0 = blockIdx.z * DV;   // V / output column slice

  const int q0 = blockIdx.x * 64 + wid * 16;

  // Q fragments for this wave's 16 rows (held in registers for all K tiles)
  v16h qf[D / 32];
#pragma unroll
  for (int i = 0; i < D / 32; ++i)
    qf[i] = load_a_frag(Qh + (size_t)q0 * ldq + i * 32, ldq);

  float m[8], l[8];
#pragma unroll
  for (int c = 0; c < 8; ++c) { m[c] = -1e30f; l[c] = 0.f; }
  v8f acc[DV / 16];
#pragma unroll
  for (int t = 0; t < DV / 16; ++t) acc[t] = zero8();

  for (int k0 = 0; k0 < S_TOK; k0 += 32) {
    __syncthreads();
    // cooperative K/V tile load into LDS (vectorized 16B)
    for (int idx = tid; idx < 32 * D / 8; idx += 128) {
      const int r = idx / (D / 8), ch = idx % (D / 8);
      *(uint4*)&lds_k[r * D + ch * 8] =
          *(const uint4*)&Kh[(size_t)(k0 + r) * ldk + ch * 8];
    }
    for (int idx = tid; idx < 32 * DV / 8; idx += 128) {
      const int r = idx / (DV / 8), ch = idx % (DV / 8);
      *(uint4*)&lds_v[r * DV + ch * 8] =
          *(const uint4*)&Vh[(size_t)(k0 + r) * ldv + vcol0 + ch * 8];
    }
    // prefetch next tile's rows into cache (global_prefetch_b8)
    if (tid < 32 && k0 + 32 < S_TOK) {
      __builtin_prefetch(&Kh[(size_t)(k0 + 32 + tid) * ldk], 0, 1);
      __builtin_prefetch(&Vh[(size_t)(k0 + 32 + tid) * ldv + vcol0], 0, 1);
    }
    __syncthreads();

    // scores S = Q * K^T for 16 rows x 32 keys (two 16-col WMMA tiles)
    v8f s0 = zero8(), s1 = zero8();
#pragma unroll
    for (int i = 0; i < D / 32; ++i) {
      v16h b0 = load_b_frag_nk(lds_k + 0 * D + i * 32, D);
      v16h b1 = load_b_frag_nk(lds_k + 16 * (size_t)D + i * 32, D);
      s0 = wmma_f16(qf[i], b0, s0);
      s1 = wmma_f16(qf[i], b1, s1);
    }

    // online softmax: in the f32 C-layout, component c of lanes {hi*16..+15}
    // holds row (c + 8*hi); its 16 columns live across those 16 lanes, so
    // xor-butterflies with masks 1/2/4/8 reduce a row within its half-wave.
    float p0[8], p1[8], sc[8];
#pragma unroll
    for (int c = 0; c < 8; ++c) {
      const float a = s0[c] * scale;
      const float b = s1[c] * scale;
      float mx = fmaxf(a, b);
#pragma unroll
      for (int msk = 1; msk <= 8; msk <<= 1)
        mx = fmaxf(mx, __shfl_xor(mx, msk, 32));
      const float mn = fmaxf(m[c], mx);
      sc[c] = __expf(m[c] - mn);
      p0[c] = __expf(a - mn);
      p1[c] = __expf(b - mn);
      float rs = p0[c] + p1[c];
#pragma unroll
      for (int msk = 1; msk <= 8; msk <<= 1) rs += __shfl_xor(rs, msk, 32);
      l[c] = l[c] * sc[c] + rs;
      m[c] = mn;
    }
#pragma unroll
    for (int t = 0; t < DV / 16; ++t)
#pragma unroll
      for (int c = 0; c < 8; ++c) acc[t][c] *= sc[c];

    // re-layout P (C-layout -> A-layout) through the per-wave LDS tile
#pragma unroll
    for (int c = 0; c < 8; ++c) {
      const int r = c + 8 * hi, col = lane & 15;
      lds_p[wid][r * 32 + col]      = (f16)p0[c];
      lds_p[wid][r * 32 + 16 + col] = (f16)p1[c];
    }
    const v16h pa = load_a_frag(&lds_p[wid][0], 32);
#pragma unroll
    for (int t = 0; t < DV / 16; ++t) {
      v16h bv = load_b_frag_kn(lds_v + t * 16, DV);
      acc[t] = wmma_f16(pa, bv, acc[t]);
    }
  }

  float* O32 = Of32 ? Of32 + (size_t)blockIdx.y * hsO : nullptr;
  f16*   O16 = Of16 ? Of16 + (size_t)blockIdx.y * hsO : nullptr;
#pragma unroll
  for (int t = 0; t < DV / 16; ++t)
#pragma unroll
    for (int c = 0; c < 8; ++c) {
      const int row = q0 + c + 8 * hi;
      const int col = vcol0 + t * 16 + (lane & 15);
      const float ov = acc[t][c] / l[c];
      if (O32) O32[(size_t)row * ldo + col] = ov;
      if (O16) O16[(size_t)row * ldo + col] = (f16)ov;
    }
}

// ---------------------------------------------------------------------------
// WMMA GEMM: Y[M,N] = epi(X[M,K] * W[N,K]^T + bias).  64x64 tile per WG
// (4 waves, 32x32 each), K-step 32.  EPI: 0=none, 1=exact GELU, 2=residual.
// ---------------------------------------------------------------------------
template <int EPI>
__global__ void __launch_bounds__(128, 2)
gemm_kernel(const f16* __restrict__ X, const f16* __restrict__ Wt,
            const float* __restrict__ bias, const float* __restrict__ res,
            float* __restrict__ Yf32, f16* __restrict__ Yf16,
            int M, int N, int Kd) {
  __shared__ __align__(16) f16 xs[64 * 32];
  __shared__ __align__(16) f16 wsl[64 * 32];

  const int tid  = threadIdx.x;
  const int wid  = tid >> 5;
  const int lane = tid & 31;
  const int hi   = lane >> 4;
  const int m0   = blockIdx.y * 64;
  const int n0   = blockIdx.x * 64;
  const int wm   = wid & 1;
  const int wn   = wid >> 1;

  v8f acc[2][2];
#pragma unroll
  for (int i = 0; i < 2; ++i)
#pragma unroll
    for (int j = 0; j < 2; ++j) acc[i][j] = zero8();

  for (int k0 = 0; k0 < Kd; k0 += 32) {
    __syncthreads();
    for (int idx = tid; idx < 64 * 32 / 8; idx += 128) {
      const int r = idx >> 2, ch = idx & 3;
      *(uint4*)&xs[r * 32 + ch * 8] =
          *(const uint4*)&X[(size_t)(m0 + r) * Kd + k0 + ch * 8];
      *(uint4*)&wsl[r * 32 + ch * 8] =
          *(const uint4*)&Wt[(size_t)(n0 + r) * Kd + k0 + ch * 8];
    }
    // prefetch next K-step tiles (global_prefetch_b8)
    if (tid < 64 && k0 + 32 < Kd) {
      __builtin_prefetch(&X[(size_t)(m0 + tid) * Kd + k0 + 32], 0, 1);
      __builtin_prefetch(&Wt[(size_t)(n0 + tid) * Kd + k0 + 32], 0, 1);
    }
    __syncthreads();

    const v16h a0 = load_a_frag(xs + (wm * 32 + 0) * 32, 32);
    const v16h a1 = load_a_frag(xs + (wm * 32 + 16) * 32, 32);
    const v16h b0 = load_b_frag_nk(wsl + (wn * 32 + 0) * 32, 32);
    const v16h b1 = load_b_frag_nk(wsl + (wn * 32 + 16) * 32, 32);
    acc[0][0] = wmma_f16(a0, b0, acc[0][0]);
    acc[0][1] = wmma_f16(a0, b1, acc[0][1]);
    acc[1][0] = wmma_f16(a1, b0, acc[1][0]);
    acc[1][1] = wmma_f16(a1, b1, acc[1][1]);
  }

#pragma unroll
  for (int i = 0; i < 2; ++i)
#pragma unroll
    for (int j = 0; j < 2; ++j)
#pragma unroll
      for (int c = 0; c < 8; ++c) {
        const int row = m0 + wm * 32 + i * 16 + c + 8 * hi;
        const int col = n0 + wn * 32 + j * 16 + (lane & 15);
        float y = acc[i][j][c] + bias[col];
        if (EPI == 1) y = 0.5f * y * (1.f + erff(y * 0.70710678118654752f));
        if (EPI == 2) y += res[(size_t)row * N + col];
        if (Yf32) Yf32[(size_t)row * N + col] = y;
        if (Yf16) Yf16[(size_t)row * N + col] = (f16)y;
      }
}

// ---------------------------------------------------------------------------
// Host orchestration
// ---------------------------------------------------------------------------
extern "C" void kernel_launch(void* const* d_in, const int* in_sizes, int n_in,
                              void* d_out, int out_size, void* d_ws, size_t ws_size,
                              hipStream_t stream) {
  (void)in_sizes; (void)n_in; (void)out_size; (void)ws_size;

  const float* ml_c    = (const float*)d_in[0];
  const float* ml_r    = (const float*)d_in[1];
  const float* fl_r    = (const float*)d_in[2];
  const float* ca_qpos = (const float*)d_in[3];
  const float* ca_kpos = (const float*)d_in[4];

  char* wsp = (char*)d_ws;
  size_t off = 0;
  auto alloc = [&](size_t bytes) -> void* {
    void* p = wsp + off;
    off = (off + bytes + 255) & ~(size_t)255;
    return p;
  };

  f16* q   = (f16*)alloc((size_t)S_TOK * DQDIM * sizeof(f16));
  f16* k   = (f16*)alloc((size_t)S_TOK * DQDIM * sizeof(f16));
  f16* v   = (f16*)alloc((size_t)S_TOK * CDIM * sizeof(f16));
  float* x = (float*)alloc((size_t)S_TOK * CDIM * sizeof(float));
  f16* xq  = (f16*)alloc((size_t)S_TOK * CDIM * sizeof(f16));
  f16* xk  = (f16*)alloc((size_t)S_TOK * CDIM * sizeof(f16));
  f16* xv  = (f16*)alloc((size_t)S_TOK * CDIM * sizeof(f16));
  f16* qh  = (f16*)alloc((size_t)S_TOK * CDIM * sizeof(f16));
  f16* kh  = (f16*)alloc((size_t)S_TOK * CDIM * sizeof(f16));
  f16* vh  = (f16*)alloc((size_t)S_TOK * CDIM * sizeof(f16));
  f16* o   = (f16*)alloc((size_t)S_TOK * CDIM * sizeof(f16));
  f16* xn2 = (f16*)alloc((size_t)S_TOK * CDIM * sizeof(f16));
  f16* h   = (f16*)alloc((size_t)S_TOK * MLPDIM * sizeof(f16));
  f16* ipw16 = (f16*)alloc((size_t)3 * CDIM * CDIM * sizeof(f16));
  f16* opw16 = (f16*)alloc((size_t)CDIM * CDIM * sizeof(f16));
  f16* w116  = (f16*)alloc((size_t)MLPDIM * CDIM * sizeof(f16));
  f16* w216  = (f16*)alloc((size_t)CDIM * MLPDIM * sizeof(f16));

  const float crossScale = 1.f / 16.f;                 // DQ^-0.5
  const float headScale  = 0.17677669529663687f;       // DH^-0.5

  // 1) build q,k,v (transpose + pos embed, f32 -> f16)
  build_qkv_kernel<<<S_TOK, 256, 0, stream>>>(ml_c, ml_r, fl_r, ca_qpos, ca_kpos,
                                              q, k, v);

  // 2) dense cross attention -> x (f32 residual stream).  DV split over
  //    grid.z (2 x 128 cols) keeps accumulators inside the VGPR budget.
  flash_attn_kernel<DQDIM, 128><<<dim3(S_TOK / 64, 1, 2), 128, 0, stream>>>(
      q, k, v, x, nullptr, DQDIM, DQDIM, CDIM, CDIM, 0, 0, 0, 0, crossScale);

  // 3) transformer blocks
  for (int blk = 0; blk < 2; ++blk) {
    const int base = 5 + 14 * blk;
    const float* n1w = (const float*)d_in[base + 0];
    const float* n1b = (const float*)d_in[base + 1];
    const float* n2w = (const float*)d_in[base + 2];
    const float* n2b = (const float*)d_in[base + 3];
    const float* ipw = (const float*)d_in[base + 4];
    const float* ipb = (const float*)d_in[base + 5];
    const float* opw = (const float*)d_in[base + 6];
    const float* opb = (const float*)d_in[base + 7];
    const float* w1  = (const float*)d_in[base + 8];
    const float* b1  = (const float*)d_in[base + 9];
    const float* w2  = (const float*)d_in[base + 10];
    const float* b2  = (const float*)d_in[base + 11];
    const float* qp  = (const float*)d_in[base + 12];
    const float* kp  = (const float*)d_in[base + 13];

    // weights f32 -> f16
    f32_to_f16_kernel<<<(3 * CDIM * CDIM + 255) / 256, 256, 0, stream>>>(
        ipw, ipw16, 3 * CDIM * CDIM);
    f32_to_f16_kernel<<<(CDIM * CDIM + 255) / 256, 256, 0, stream>>>(
        opw, opw16, CDIM * CDIM);
    f32_to_f16_kernel<<<(MLPDIM * CDIM + 255) / 256, 256, 0, stream>>>(
        w1, w116, MLPDIM * CDIM);
    f32_to_f16_kernel<<<(CDIM * MLPDIM + 255) / 256, 256, 0, stream>>>(
        w2, w216, CDIM * MLPDIM);

    // LN1 + pos embeds -> f16 GEMM inputs
    ln_kernel<<<S_TOK, 256, 0, stream>>>(x, n1w, n1b, qp, kp, xq, xk, xv);

    // in-projections (each [S,256] x [256,256]^T)
    gemm_kernel<0><<<dim3(CDIM / 64, S_TOK / 64), 128, 0, stream>>>(
        xq, ipw16, ipb, nullptr, nullptr, qh, S_TOK, CDIM, CDIM);
    gemm_kernel<0><<<dim3(CDIM / 64, S_TOK / 64), 128, 0, stream>>>(
        xk, ipw16 + (size_t)CDIM * CDIM, ipb + CDIM, nullptr, nullptr, kh,
        S_TOK, CDIM, CDIM);
    gemm_kernel<0><<<dim3(CDIM / 64, S_TOK / 64), 128, 0, stream>>>(
        xv, ipw16 + (size_t)2 * CDIM * CDIM, ipb + 2 * CDIM, nullptr, nullptr,
        vh, S_TOK, CDIM, CDIM);

    // per-head attention (8 heads of dim 32), output f16 for out-projection
    flash_attn_kernel<DHEAD, DHEAD><<<dim3(S_TOK / 64, NHEADS, 1), 128, 0, stream>>>(
        qh, kh, vh, nullptr, o, CDIM, CDIM, CDIM, CDIM, DHEAD, DHEAD, DHEAD,
        DHEAD, headScale);

    // x = x + o @ opw^T + opb (residual epilogue, in-place on x)
    gemm_kernel<2><<<dim3(CDIM / 64, S_TOK / 64), 128, 0, stream>>>(
        o, opw16, opb, x, x, nullptr, S_TOK, CDIM, CDIM);

    // LN2 -> f16
    ln_kernel<<<S_TOK, 256, 0, stream>>>(x, n2w, n2b, nullptr, nullptr, nullptr,
                                         nullptr, xn2);

    // h = gelu(xn2 @ w1^T + b1)   [S,1024]
    gemm_kernel<1><<<dim3(MLPDIM / 64, S_TOK / 64), 128, 0, stream>>>(
        xn2, w116, b1, nullptr, nullptr, h, S_TOK, MLPDIM, CDIM);

    // x = x + h @ w2^T + b2
    gemm_kernel<2><<<dim3(CDIM / 64, S_TOK / 64), 128, 0, stream>>>(
        h, w216, b2, x, x, nullptr, S_TOK, CDIM, MLPDIM);
  }

  // 4) [S,C] -> [C,H,W]
  transpose_out_kernel<<<S_TOK, 256, 0, stream>>>(x, (float*)d_out);
}
